// MultiheadAttention2d_7009386627193
// MI455X (gfx1250) — matrix operand order repaired and checked by reading, not compiled, not run on hardware
//
#include <hip/hip_runtime.h>
#include <hip/hip_bf16.h>

typedef _Float16 v16h __attribute__((ext_vector_type(16)));
typedef float    v8f  __attribute__((ext_vector_type(8)));

#define NH     8
#define KS     64
#define C_TOT  1536
#define HW     1024
#define NG     24
#define BQ     128    // queries per block
#define BK     64     // keys per tile
#define PITCH  68     // LDS row pitch in halves (64 + 4 pad)

union FragU { v16h h; unsigned int u[8]; };

// A-matrix 16x32 f16 fragment (M rows = lane%16, K along columns, contiguous in LDS)
__device__ __forceinline__ v16h load_frag_A(const _Float16* base, int m0, int k0,
                                            int l16, int hi) {
    FragU f;
#pragma unroll
    for (int j = 0; j < 8; ++j) {
        int kc = ((j >> 2) << 4) + (hi << 3) + ((j & 3) << 1);   // K pair base
        f.u[j] = *(const unsigned int*)(base + (m0 + l16) * PITCH + k0 + kc);
    }
    return f.h;
}

// B-matrix 32x16 f16 fragment (N = lane%16 selects LDS row, K contiguous in row)
__device__ __forceinline__ v16h load_frag_B(const _Float16* base, int n0, int k0,
                                            int l16, int hi) {
    FragU f;
#pragma unroll
    for (int j = 0; j < 8; ++j) {
        f.u[j] = *(const unsigned int*)(base + (n0 + l16) * PITCH + k0 + ((j >> 2) << 4) + (hi << 3) + ((j & 3) << 1));
    }
    return f.h;
}

// ---------------- Kernel 1: GroupNorm statistics -----------------------------
// One block per (b, group): 64 channels x 1024 spatial = 65536 floats, contiguous.
__global__ __launch_bounds__(256) void gn_stats_kernel(const float* __restrict__ x,
                                                       float* __restrict__ stats) {
    const int bg  = blockIdx.x;                 // 0..191 == b*24+g
    const int tid = threadIdx.x;
    const float4* p = (const float4*)(x + (size_t)bg * 64 * HW);   // 16384 float4
    float s = 0.f, ss = 0.f;
#pragma unroll 8
    for (int i = 0; i < 64; ++i) {
        float4 v = p[tid + i * 256];
        s  += v.x + v.y + v.z + v.w;
        ss += v.x * v.x + v.y * v.y + v.z * v.z + v.w * v.w;
    }
#pragma unroll
    for (int off = 16; off > 0; off >>= 1) {
        s  += __shfl_xor(s,  off, 32);
        ss += __shfl_xor(ss, off, 32);
    }
    __shared__ float rs[8], rss[8];
    const int w = tid >> 5, ln = tid & 31;
    if (ln == 0) { rs[w] = s; rss[w] = ss; }
    __syncthreads();
    if (tid == 0) {
        float S = 0.f, SS = 0.f;
#pragma unroll
        for (int i = 0; i < 8; ++i) { S += rs[i]; SS += rss[i]; }
        float mean = S * (1.f / 65536.f);
        float var  = SS * (1.f / 65536.f) - mean * mean;
        stats[2 * bg]     = mean;
        stats[2 * bg + 1] = rsqrtf(var + 1e-5f);
    }
}

// ---------------- Kernel 2: fused GN + flash attention -----------------------
__global__ __launch_bounds__(256) void attn_kernel(const float* __restrict__ x,
                                                   const float* __restrict__ gw,
                                                   const float* __restrict__ gb,
                                                   const float* __restrict__ stats,
                                                   float* __restrict__ out) {
    __shared__ _Float16 Qlds[BQ][PITCH];      // [q][c]
    __shared__ _Float16 Klds[BK][PITCH];      // [k][c]
    __shared__ _Float16 Vlds[KS][PITCH];      // [v][k]
    __shared__ _Float16 Plds[8][16][PITCH];   // per-wave P scratch [q][k]

    const int qt = blockIdx.x, h = blockIdx.y, b = blockIdx.z;
    const int tid = threadIdx.x;
    const int w = tid >> 5, lane = tid & 31, hi = lane >> 4, l16 = lane & 15;

    const int ch0 = h * 192;
    const size_t xbase = ((size_t)b * C_TOT + ch0) * HW;

    const int sg = (b * NG + h * 3) * 2;
    const float meanK = stats[sg + 0], rstdK = stats[sg + 1];
    const float meanQ = stats[sg + 2], rstdQ = stats[sg + 3];
    const float meanV = stats[sg + 4], rstdV = stats[sg + 5];

    // ---- stage Q tile (64c x 128q) transposed into Qlds[q][c], GN fused
    {
        const float* qp = x + xbase + (size_t)KS * HW;
        const int q0 = qt * BQ;
#pragma unroll
        for (int i = 0; i < 8; ++i) {
            int idx = tid + i * 256;          // 0..2047
            int c   = idx >> 5;               // 0..63
            int qd  = idx & 31;               // float4 index along q
            float4 v = *(const float4*)(qp + (size_t)c * HW + q0 + qd * 4);
            float g = gw[ch0 + KS + c], be = gb[ch0 + KS + c];
            Qlds[qd * 4 + 0][c] = (_Float16)((v.x - meanQ) * rstdQ * g + be);
            Qlds[qd * 4 + 1][c] = (_Float16)((v.y - meanQ) * rstdQ * g + be);
            Qlds[qd * 4 + 2][c] = (_Float16)((v.z - meanQ) * rstdQ * g + be);
            Qlds[qd * 4 + 3][c] = (_Float16)((v.w - meanQ) * rstdQ * g + be);
        }
    }
    __syncthreads();

    // Q fragments stay resident across the whole K loop
    const v16h aq0 = load_frag_A(&Qlds[0][0], w * 16, 0,  l16, hi);
    const v16h aq1 = load_frag_A(&Qlds[0][0], w * 16, 32, l16, hi);

    v8f oacc[4];
    float m_r[8], l_r[8];
#pragma unroll
    for (int t = 0; t < 4; ++t)
#pragma unroll
        for (int r = 0; r < 8; ++r) oacc[t][r] = 0.f;
#pragma unroll
    for (int r = 0; r < 8; ++r) { m_r[r] = -1e30f; l_r[r] = 0.f; }

    for (int kt = 0; kt < HW / BK; ++kt) {
        __syncthreads();   // previous tile fully consumed
        // ---- stage K tile (transposed -> Klds[k][c]) and V tile (Vlds[v][k])
        {
            const float* kp = x + xbase;
            const float* vp = x + xbase + (size_t)(2 * KS) * HW;
            const int s0 = kt * BK;
#pragma unroll
            for (int i = 0; i < 4; ++i) {
                int idx = tid + i * 256;     // 0..1023
                int c   = idx >> 4;          // 0..63
                int qd  = idx & 15;          // float4 index along spatial
                {
                    float4 v = *(const float4*)(kp + (size_t)c * HW + s0 + qd * 4);
                    float g = gw[ch0 + c], be = gb[ch0 + c];
                    Klds[qd * 4 + 0][c] = (_Float16)((v.x - meanK) * rstdK * g + be);
                    Klds[qd * 4 + 1][c] = (_Float16)((v.y - meanK) * rstdK * g + be);
                    Klds[qd * 4 + 2][c] = (_Float16)((v.z - meanK) * rstdK * g + be);
                    Klds[qd * 4 + 3][c] = (_Float16)((v.w - meanK) * rstdK * g + be);
                }
                {
                    float4 v = *(const float4*)(vp + (size_t)c * HW + s0 + qd * 4);
                    float g = gw[ch0 + 2 * KS + c], be = gb[ch0 + 2 * KS + c];
                    Vlds[c][qd * 4 + 0] = (_Float16)((v.x - meanV) * rstdV * g + be);
                    Vlds[c][qd * 4 + 1] = (_Float16)((v.y - meanV) * rstdV * g + be);
                    Vlds[c][qd * 4 + 2] = (_Float16)((v.z - meanV) * rstdV * g + be);
                    Vlds[c][qd * 4 + 3] = (_Float16)((v.w - meanV) * rstdV * g + be);
                }
            }
        }
        __syncthreads();

        // ---- S = Q^T K  (16 q-rows x 64 k-cols)
        v8f sacc[4];
#pragma unroll
        for (int t = 0; t < 4; ++t)
#pragma unroll
            for (int r = 0; r < 8; ++r) sacc[t][r] = 0.f;
#pragma unroll
        for (int nt = 0; nt < 4; ++nt) {
            v16h b0 = load_frag_B(&Klds[0][0], nt * 16, 0,  l16, hi);
            v16h b1 = load_frag_B(&Klds[0][0], nt * 16, 32, l16, hi);
            sacc[nt] = __builtin_amdgcn_wmma_f32_16x16x32_f16(false, aq0, false, b0,
                                                              (short)0, sacc[nt], false, false);
            sacc[nt] = __builtin_amdgcn_wmma_f32_16x16x32_f16(false, aq1, false, b1,
                                                              (short)0, sacc[nt], false, false);
        }

        // ---- online softmax (rows live in one 16-lane half: butterfly width 16)
#pragma unroll
        for (int r = 0; r < 8; ++r) {
            float mx = -1e30f;
#pragma unroll
            for (int nt = 0; nt < 4; ++nt) {
                sacc[nt][r] *= 0.125f;                    // 1/sqrt(64)
                mx = fmaxf(mx, sacc[nt][r]);
            }
#pragma unroll
            for (int off = 8; off >= 1; off >>= 1) mx = fmaxf(mx, __shfl_xor(mx, off, 32));
            float nm   = fmaxf(m_r[r], mx);
            float corr = __expf(m_r[r] - nm);
            m_r[r] = nm;
            float rsum = 0.f;
#pragma unroll
            for (int nt = 0; nt < 4; ++nt) {
                float pv = __expf(sacc[nt][r] - nm);
                sacc[nt][r] = pv;
                rsum += pv;
            }
#pragma unroll
            for (int off = 8; off >= 1; off >>= 1) rsum += __shfl_xor(rsum, off, 32);
            l_r[r] = l_r[r] * corr + rsum;
#pragma unroll
            for (int vt = 0; vt < 4; ++vt) oacc[vt][r] *= corr;
        }

        // ---- P (C-layout) -> per-wave LDS scratch (A-layout source)
#pragma unroll
        for (int nt = 0; nt < 4; ++nt)
#pragma unroll
            for (int r = 0; r < 8; ++r)
                Plds[w][r + hi * 8][nt * 16 + l16] = (_Float16)sacc[nt][r];

        asm volatile("s_wait_dscnt 0" ::: "memory");   // same-wave LDS RAW fence

        // ---- O += P * V^T
#pragma unroll
        for (int cc = 0; cc < 2; ++cc) {
            v16h ap = load_frag_A(&Plds[w][0][0], 0, cc * 32, l16, hi);
#pragma unroll
            for (int vt = 0; vt < 4; ++vt) {
                v16h bv = load_frag_B(&Vlds[0][0], vt * 16, cc * 32, l16, hi);
                oacc[vt] = __builtin_amdgcn_wmma_f32_16x16x32_f16(false, ap, false, bv,
                                                                  (short)0, oacc[vt], false, false);
            }
        }
    }

    // ---- epilogue: normalize by softmax denominator and store
    const int qabs_base = qt * BQ + w * 16 + hi * 8;
#pragma unroll
    for (int r = 0; r < 8; ++r) {
        float inv = 1.f / l_r[r];
        int q = qabs_base + r;
#pragma unroll
        for (int vt = 0; vt < 4; ++vt) {
            int vch = h * KS + vt * 16 + l16;
            out[((size_t)b * (NH * KS) + vch) * HW + q] = oacc[vt][r] * inv;
        }
    }
}

extern "C" void kernel_launch(void* const* d_in, const int* in_sizes, int n_in,
                              void* d_out, int out_size, void* d_ws, size_t ws_size,
                              hipStream_t stream) {
    const float* x  = (const float*)d_in[0];
    const float* gw = (const float*)d_in[1];
    const float* gb = (const float*)d_in[2];
    float* out   = (float*)d_out;
    float* stats = (float*)d_ws;   // 192 * 2 floats

    gn_stats_kernel<<<8 * NG, 256, 0, stream>>>(x, stats);
    dim3 grid(HW / BQ, NH, 8);     // (8 q-tiles, 8 heads, 8 batches) = 512 blocks
    attn_kernel<<<grid, 256, 0, stream>>>(x, gw, gb, stats, out);
}
